// TransformerEncoderLayer_55241869361203
// MI455X (gfx1250) — matrix operand order repaired and checked by reading, not compiled, not run on hardware
//
#include <hip/hip_runtime.h>
#include <hip/hip_bf16.h>
#include <math.h>

// ---------------- problem constants ----------------
#define C_DIM  256
#define N_POS  1024
#define NHEAD  8
#define HDIM   32
#define F_DIM  2048
#define BATCH  8

// ---------------- tiling constants -----------------
#define APITCH 40    // LDS pitch (ushorts) for 128x32 A tiles: 64B row + 16B pad
#define BPITCH 40    // LDS pitch (ushorts) for 128x32 B tiles
#define SPITCH 136   // LDS pitch for attention S/P/V tiles: 256B row + 16B pad

typedef unsigned short ushort_t;
typedef __attribute__((ext_vector_type(16))) __bf16 v16bf;
typedef __attribute__((ext_vector_type(8)))  float  v8f;
typedef __attribute__((ext_vector_type(4)))  unsigned int u32x4;
typedef __attribute__((ext_vector_type(8)))  unsigned int u32x8;

union FragU { unsigned int u[8]; v16bf v; };

static __device__ __forceinline__ ushort_t f2bf(float f) {
  unsigned int u = __float_as_uint(f);
  unsigned int r = (u + 0x7FFFu + ((u >> 16) & 1u)) >> 16;   // RNE
  return (ushort_t)r;
}
static __device__ __forceinline__ float bf2f(ushort_t h) {
  return __uint_as_float(((unsigned int)h) << 16);
}

// ---------------- CDNA5 async data movement ----------------
// per-lane async copy of 16B global -> LDS (ASYNCcnt)
static __device__ __forceinline__ void async_cp16(ushort_t* ldsDst, const ushort_t* gSrc) {
  unsigned l = (unsigned)(uintptr_t)ldsDst;   // LDS aperture: low 32 bits = LDS byte offset
  asm volatile("global_load_async_to_lds_b128 %0, %1, off"
               :: "v"(l), "v"(gSrc) : "memory");
}
static __device__ __forceinline__ void wait_async() {
#if __has_builtin(__builtin_amdgcn_s_wait_asynccnt)
  __builtin_amdgcn_s_wait_asynccnt(0);
#else
  asm volatile("s_wait_asynccnt 0x0" ::: "memory");
#endif
}
static __device__ __forceinline__ void wait_tensor() {
#if __has_builtin(__builtin_amdgcn_s_wait_tensorcnt)
  __builtin_amdgcn_s_wait_tensorcnt(0);
#else
  asm volatile("s_wait_tensorcnt 0x0" ::: "memory");
#endif
}

// TDM tile load: [rows] x [tile_k] bf16 tile, row stride = stride elems, into
// pitched LDS (pad_interval/pad_amount codes per ISA 8.4). Issue wave-uniform.
static __device__ __forceinline__ void tdm_load_tile(ushort_t* ldsDst, const ushort_t* gSrc,
                                                     unsigned tile_k, unsigned rows,
                                                     unsigned stride, unsigned pad_iv_code,
                                                     unsigned pad_amt_code) {
  unsigned long long ga = (unsigned long long)(uintptr_t)gSrc;
  u32x4 g0;
  u32x8 g1;
  g0[0] = 1u;                                     // count=1 (valid user descriptor)
  g0[1] = (unsigned)(uintptr_t)ldsDst;            // lds_addr (bytes)
  g0[2] = (unsigned)ga;                           // global_addr[31:0]
  g0[3] = (unsigned)(ga >> 32) | 0x80000000u;     // global_addr[56:32] | type=2
  g1[0] = (1u << 16) | (1u << 20) | (pad_iv_code << 22) | (pad_amt_code << 25);
  g1[1] = 0u;                                     // barrier addr 0; tensor_dim0[15:0]=0
  g1[2] = 0x0010u;                                // tensor_dim0 = 1<<20 (no clip); dim1[15:0]=0
  g1[3] = 0x0010u | (tile_k << 16);               // tensor_dim1 = 1<<20; tile_dim0
  g1[4] = rows;                                   // tile_dim1 = rows (tile_dim2 = 0)
  g1[5] = stride;                                 // tensor_dim0_stride[31:0]
  g1[6] = 0u;
  g1[7] = 0u;
  asm volatile("tensor_load_to_lds %0, %1" :: "s"(g0), "s"(g1) : "memory");
}

// ---- WMMA fragment builders (ISA 7.12.2 16-bit layouts, wave32) ----
static __device__ __forceinline__ v16bf load_a_frag(const ushort_t* lds, int row0,
                                                    int pitch, int k0, int lane) {
  FragU r;
  const ushort_t* base = lds + (row0 + (lane & 15)) * pitch + k0;
  const int hi = (lane >= 16);
#pragma unroll
  for (int v = 0; v < 8; ++v) {
    int kb = (v < 4) ? ((hi ? 8 : 0) + 2 * v) : ((hi ? 24 : 16) + 2 * (v - 4));
    r.u[v] = *(const unsigned int*)(base + kb);
  }
  return r.v;
}
static __device__ __forceinline__ v16bf load_b_frag(const ushort_t* lds, int n0,
                                                    int pitch, int k0, int lane) {
  FragU r;
  const ushort_t* base = lds + (n0 + (lane & 15)) * pitch + k0;
  const int hi = (lane >= 16);
#pragma unroll
  for (int v = 0; v < 8; ++v) {
    int kb = (v < 4) ? ((hi ? 8 : 0) + 2 * v) : ((hi ? 24 : 16) + 2 * (v - 4));
    r.u[v] = *(const unsigned int*)(base + kb);
  }
  return r.v;
}

// 32x64 wave tile: 2 A frags x 4 B frags -> 8 WMMAs (0.75 frag-loads/WMMA)
static __device__ __forceinline__ void mma32x64(const ushort_t* ldsA, int arow,
                                                const ushort_t* ldsB, int bn0,
                                                v8f acc[2][4], int lane) {
  v16bf a0 = load_a_frag(ldsA, arow,      APITCH, 0, lane);
  v16bf a1 = load_a_frag(ldsA, arow + 16, APITCH, 0, lane);
#pragma unroll
  for (int t = 0; t < 4; ++t) {
    v16bf b = load_b_frag(ldsB, bn0 + 16 * t, BPITCH, 0, lane);
    acc[0][t] = __builtin_amdgcn_wmma_f32_16x16x32_bf16(false, a0, false, b,
                                                        (short)0, acc[0][t], false, false);
    acc[1][t] = __builtin_amdgcn_wmma_f32_16x16x32_bf16(false, a1, false, b,
                                                        (short)0, acc[1][t], false, false);
  }
}

// =====================================================================
// weight conversion / repack
// =====================================================================
__global__ void cvt_bf16_kernel(const float* __restrict__ in, ushort_t* __restrict__ out, int n) {
  int i = blockIdx.x * 256 + threadIdx.x;
  if (i < n) out[i] = f2bf(in[i]);
}

// conv1_w [F][C][3][3] f32 -> w1r [tap][F][C] bf16
__global__ void repack_conv1_kernel(const float* __restrict__ w1, ushort_t* __restrict__ w1r) {
  int i = blockIdx.x * 256 + threadIdx.x;   // over F*C
  if (i < F_DIM * C_DIM) {
    int f = i >> 8, c = i & 255;
#pragma unroll
    for (int tap = 0; tap < 9; ++tap)
      w1r[((size_t)tap * F_DIM + f) * C_DIM + c] = f2bf(w1[((size_t)f * C_DIM + c) * 9 + tap]);
  }
}

// =====================================================================
// pixnorm + positional add: produce q/k/v GEMM inputs, channel-last bf16
// =====================================================================
__global__ void norm_qkv_kernel(const float* __restrict__ src, const float* __restrict__ src1,
                                const float* __restrict__ pos, const float* __restrict__ alpha1,
                                ushort_t* __restrict__ qin, ushort_t* __restrict__ kin,
                                ushort_t* __restrict__ vin) {
  __shared__ float r0[256], r1[256];
  int b = blockIdx.x >> 10, p = blockIdx.x & 1023, c = threadIdx.x;
  float s0 = src [((size_t)b * C_DIM + c) * N_POS + p];
  float s1 = src1[((size_t)b * C_DIM + c) * N_POS + p];
  r0[c] = s0 * s0; r1[c] = s1 * s1;
  __syncthreads();
  for (int s = 128; s > 0; s >>= 1) {
    if (c < s) { r0[c] += r0[c + s]; r1[c] += r1[c + s]; }
    __syncthreads();
  }
  float rs0 = rsqrtf(r0[0] * (1.0f / C_DIM) + 1e-8f);
  float rs1 = rsqrtf(r1[0] * (1.0f / C_DIM) + 1e-8f);
  float a1 = alpha1[c];
  size_t o = ((size_t)b * N_POS + p) * C_DIM + c;
  float po = pos[o];
  float n0 = s0 * rs0 * a1, n1 = s1 * rs1 * a1;
  qin[o] = f2bf(n0 + po);
  kin[o] = f2bf(n1 + po);
  vin[o] = f2bf(n1);
}

// pixnorm of xres [b][c][p] f32 -> xn2 [b][p][c] bf16
__global__ void norm2_kernel(const float* __restrict__ xres, const float* __restrict__ alpha2,
                             ushort_t* __restrict__ xn2) {
  __shared__ float r0[256];
  int b = blockIdx.x >> 10, p = blockIdx.x & 1023, c = threadIdx.x;
  float s = xres[((size_t)b * C_DIM + c) * N_POS + p];
  r0[c] = s * s;
  __syncthreads();
  for (int t = 128; t > 0; t >>= 1) {
    if (c < t) r0[c] += r0[c + t];
    __syncthreads();
  }
  float rs = rsqrtf(r0[0] * (1.0f / C_DIM) + 1e-8f);
  xn2[((size_t)b * N_POS + p) * C_DIM + c] = f2bf(s * rs * alpha2[c]);
}

// =====================================================================
// projection GEMM: block tile 128x128, TDM-staged, double buffered
// MODE 0: Q -> [b][h][n][d] bf16   MODE 1: K -> same
// MODE 2: V -> Vt [b][h][d][n] bf16
// MODE 3: O -> xres[b][c][p] = src + out   (f32)
// =====================================================================
template <int MODE>
__global__ void proj_kernel(const ushort_t* __restrict__ Ag, const ushort_t* __restrict__ Wb,
                            const float* __restrict__ bias, ushort_t* __restrict__ outb,
                            float* __restrict__ outf, const float* __restrict__ resid) {
  __shared__ __align__(16) ushort_t ldsA[2][128 * APITCH];
  __shared__ __align__(16) ushort_t ldsB[2][128 * BPITCH];
  int tid = threadIdx.x, lane = tid & 31, w = tid >> 5, wm = w >> 1, wn = w & 1;
  int b = blockIdx.z;
  const ushort_t* Ab = Ag + ((size_t)b * N_POS + (size_t)blockIdx.x * 128) * C_DIM;
  const ushort_t* Bb = Wb + (size_t)blockIdx.y * 128 * C_DIM;
  const int KSTEPS = C_DIM / 32;
  if (w == 0) {
    tdm_load_tile(ldsA[0], Ab, 32u, 128u, C_DIM, 3u, 3u);
    tdm_load_tile(ldsB[0], Bb, 32u, 128u, C_DIM, 3u, 3u);
  }
  v8f acc[2][4] = {};
  for (int ks = 0; ks < KSTEPS; ++ks) {
    int cur = ks & 1;
    if (w == 0) wait_tensor();
    __syncthreads();
    if (ks + 1 < KSTEPS && w == 0) {
      tdm_load_tile(ldsA[cur ^ 1], Ab + (ks + 1) * 32, 32u, 128u, C_DIM, 3u, 3u);
      tdm_load_tile(ldsB[cur ^ 1], Bb + (ks + 1) * 32, 32u, 128u, C_DIM, 3u, 3u);
    }
    mma32x64(ldsA[cur], wm * 32, ldsB[cur], wn * 64, acc, lane);
  }
  int mg = blockIdx.x * 128 + wm * 32;
#pragma unroll
  for (int r = 0; r < 2; ++r) {
#pragma unroll
    for (int t = 0; t < 4; ++t) {
      int nbase = blockIdx.y * 128 + wn * 64 + t * 16 + (lane & 15);
      float bi = bias[nbase];
#pragma unroll
      for (int v = 0; v < 8; ++v) {
        int m = mg + r * 16 + v + ((lane & 16) ? 8 : 0);
        float val = acc[r][t][v] + bi;
        if (MODE == 0 || MODE == 1) {
          int h = nbase >> 5, d = nbase & 31;
          outb[(((size_t)b * NHEAD + h) * N_POS + m) * HDIM + d] = f2bf(val);
        } else if (MODE == 2) {
          int h = nbase >> 5, d = nbase & 31;
          outb[(((size_t)b * NHEAD + h) * HDIM + d) * N_POS + m] = f2bf(val);
        } else {
          size_t idx = ((size_t)b * C_DIM + nbase) * N_POS + m;
          outf[idx] = resid[idx] + val;
        }
      }
    }
  }
}

// =====================================================================
// flash-style attention, TDM-staged K/V tiles
// =====================================================================
__global__ void attn_kernel(const ushort_t* __restrict__ Q, const ushort_t* __restrict__ K,
                            const ushort_t* __restrict__ Vt, ushort_t* __restrict__ attn_out) {
  __shared__ __align__(16) ushort_t ldsQ[64 * APITCH];
  __shared__ __align__(16) ushort_t ldsK[128 * BPITCH];
  __shared__ float    ldsS[64 * SPITCH];
  __shared__ __align__(16) ushort_t ldsP[64 * SPITCH];
  __shared__ __align__(16) ushort_t ldsV[32 * SPITCH];
  __shared__ float rowm[64], rowl[64], rowf[64];

  int tid = threadIdx.x, lane = tid & 31, w = tid >> 5, wm = w >> 1, wn = w & 1;
  int bh = blockIdx.y;
  const ushort_t* Qb = Q + ((size_t)bh * N_POS + (size_t)blockIdx.x * 64) * HDIM;
  const ushort_t* Kb = K + (size_t)bh * N_POS * HDIM;
  const ushort_t* Vb = Vt + (size_t)bh * HDIM * N_POS;

  // stage Q (pre-scaled by 1/sqrt(d))
  {
    int row = tid >> 2, kg = (tid & 3) * 8;
#pragma unroll
    for (int i = 0; i < 8; ++i)
      ldsQ[row * APITCH + kg + i] = f2bf(bf2f(Qb[(size_t)row * HDIM + kg + i]) * 0.17677669529f);
  }
  if (tid < 64) { rowm[tid] = -3.0e38f; rowl[tid] = 0.0f; }
  v8f oacc = {};
  __syncthreads();

  for (int jt = 0; jt < 8; ++jt) {
    // stage K tile (128 rows x 32 d, row stride 32, 64B rows)
    if (w == 0) {
      tdm_load_tile(ldsK, Kb + (size_t)jt * 128 * HDIM, 32u, 128u, HDIM, 3u, 3u);
      wait_tensor();
    }
    __syncthreads();
    // S = Q . K^T : wave tile 16x64
#pragma unroll
    for (int t = 0; t < 4; ++t) {
      v16bf a = load_a_frag(ldsQ, wm * 16, APITCH, 0, lane);
      v16bf bb = load_b_frag(ldsK, wn * 64 + t * 16, BPITCH, 0, lane);
      v8f s = {};
      s = __builtin_amdgcn_wmma_f32_16x16x32_bf16(false, a, false, bb, (short)0, s, false, false);
      int nn = wn * 64 + t * 16 + (lane & 15);
#pragma unroll
      for (int v = 0; v < 8; ++v)
        ldsS[(wm * 16 + v + ((lane & 16) ? 8 : 0)) * SPITCH + nn] = s[v];
    }
    __syncthreads();
    // online softmax (one thread per row)
    if (tid < 64) {
      float mold = rowm[tid], mx = mold;
      const float* sr = ldsS + tid * SPITCH;
      for (int j = 0; j < 128; ++j) mx = fmaxf(mx, sr[j]);
      float f = __expf(mold - mx);
      float l = rowl[tid] * f;
      ushort_t* pr = ldsP + tid * SPITCH;
      for (int j = 0; j < 128; ++j) {
        float e = __expf(sr[j] - mx);
        l += e;
        pr[j] = f2bf(e);
      }
      rowm[tid] = mx; rowl[tid] = l; rowf[tid] = f;
    }
    // stage V tile [32 d][128 j] (row stride 1024, 256B rows -> pad code 5)
    if (w == 0) {
      tdm_load_tile(ldsV, Vb + jt * 128, 128u, 32u, N_POS, 5u, 3u);
      wait_tensor();
    }
    __syncthreads();
    // rescale O accumulator, then O += P . V (K=128 -> 4 k-steps)
#pragma unroll
    for (int v = 0; v < 8; ++v)
      oacc[v] *= rowf[wm * 16 + v + ((lane & 16) ? 8 : 0)];
#pragma unroll
    for (int ks = 0; ks < 4; ++ks) {
      v16bf a = load_a_frag(ldsP, wm * 16, SPITCH, ks * 32, lane);
      v16bf bb = load_b_frag(ldsV, wn * 16, SPITCH, ks * 32, lane);
      oacc = __builtin_amdgcn_wmma_f32_16x16x32_bf16(false, a, false, bb, (short)0, oacc, false, false);
    }
    __syncthreads();
  }
  int b = bh >> 3, h = bh & 7;
#pragma unroll
  for (int v = 0; v < 8; ++v) {
    int mm = wm * 16 + v + ((lane & 16) ? 8 : 0);
    float ov = oacc[v] / rowl[mm];
    int m = blockIdx.x * 64 + mm;
    int d = wn * 16 + (lane & 15);
    attn_out[((size_t)b * N_POS + m) * C_DIM + h * HDIM + d] = f2bf(ov);
  }
}

// =====================================================================
// 3x3 conv implicit GEMM: block tile 128x128, async-LDS A (zero-padded
// borders), TDM B, double buffered; fused bias + exact GELU
// =====================================================================
__global__ void conv1_kernel(const ushort_t* __restrict__ xn2, const ushort_t* __restrict__ w1r,
                             const float* __restrict__ b1, ushort_t* __restrict__ y) {
  __shared__ __align__(16) ushort_t ldsA[2][128 * APITCH];
  __shared__ __align__(16) ushort_t ldsB[2][128 * BPITCH];
  int tid = threadIdx.x, lane = tid & 31, w = tid >> 5, wm = w >> 1, wn = w & 1;
  int b = blockIdx.z;
  const ushort_t* Xb = xn2 + (size_t)b * N_POS * C_DIM;
  const ushort_t* Wb = w1r + (size_t)blockIdx.y * 128 * C_DIM;
  const int TOT = 9 * (C_DIM / 32);

  auto stageA = [&](ushort_t* dst, int kk) {
    int tap = kk >> 3, k0 = (kk & 7) * 32;
    int ky = tap / 3 - 1, kx = tap % 3 - 1;
    int row = tid >> 1, kg = (tid & 1) * 16;
    int p = blockIdx.x * 128 + row;
    int yy = (p >> 5) + ky, xx = (p & 31) + kx;
    ushort_t* d = dst + row * APITCH + kg;
    if ((unsigned)yy < 32u && (unsigned)xx < 32u) {
      const ushort_t* s = Xb + (size_t)(yy * 32 + xx) * C_DIM + k0 + kg;
      async_cp16(d, s);
      async_cp16(d + 8, s + 8);
    } else {
      *(uint4*)d = make_uint4(0u, 0u, 0u, 0u);
      *(uint4*)(d + 8) = make_uint4(0u, 0u, 0u, 0u);
    }
  };
  auto stageB = [&](ushort_t* dst, int kk) {
    if (w == 0) {
      int tap = kk >> 3, k0 = (kk & 7) * 32;
      tdm_load_tile(dst, Wb + (size_t)tap * F_DIM * C_DIM + k0, 32u, 128u, C_DIM, 3u, 3u);
    }
  };

  stageA(ldsA[0], 0);
  stageB(ldsB[0], 0);
  v8f acc[2][4] = {};
  for (int kk = 0; kk < TOT; ++kk) {
    int cur = kk & 1;
    wait_async();
    if (w == 0) wait_tensor();
    __syncthreads();
    if (kk + 1 < TOT) {
      stageA(ldsA[cur ^ 1], kk + 1);
      stageB(ldsB[cur ^ 1], kk + 1);
    }
    mma32x64(ldsA[cur], wm * 32, ldsB[cur], wn * 64, acc, lane);
  }
  int mg = blockIdx.x * 128 + wm * 32;
#pragma unroll
  for (int r = 0; r < 2; ++r) {
#pragma unroll
    for (int t = 0; t < 4; ++t) {
      int nbase = blockIdx.y * 128 + wn * 64 + t * 16 + (lane & 15);
      float bi = b1[nbase];
#pragma unroll
      for (int v = 0; v < 8; ++v) {
        int m = mg + r * 16 + v + ((lane & 16) ? 8 : 0);
        float xg = acc[r][t][v] + bi;
        float g = 0.5f * xg * (1.0f + erff(xg * 0.70710678119f));
        y[((size_t)b * N_POS + m) * F_DIM + nbase] = f2bf(g);
      }
    }
  }
}

// =====================================================================
// 1x1 conv GEMM: block tile 128x128, TDM-staged, double buffered;
// fused bias + residual, final NCHW f32 output
// =====================================================================
__global__ void conv2_kernel(const ushort_t* __restrict__ y, const ushort_t* __restrict__ w2b,
                             const float* __restrict__ b2, const float* __restrict__ xres,
                             float* __restrict__ out) {
  __shared__ __align__(16) ushort_t ldsA[2][128 * APITCH];
  __shared__ __align__(16) ushort_t ldsB[2][128 * BPITCH];
  int tid = threadIdx.x, lane = tid & 31, w = tid >> 5, wm = w >> 1, wn = w & 1;
  int b = blockIdx.z;
  const ushort_t* Ab = y + ((size_t)b * N_POS + (size_t)blockIdx.x * 128) * F_DIM;
  const ushort_t* Bb = w2b + (size_t)blockIdx.y * 128 * F_DIM;
  const int KSTEPS = F_DIM / 32;
  if (w == 0) {
    tdm_load_tile(ldsA[0], Ab, 32u, 128u, F_DIM, 3u, 3u);
    tdm_load_tile(ldsB[0], Bb, 32u, 128u, F_DIM, 3u, 3u);
  }
  v8f acc[2][4] = {};
  for (int ks = 0; ks < KSTEPS; ++ks) {
    int cur = ks & 1;
    if (w == 0) wait_tensor();
    __syncthreads();
    if (ks + 1 < KSTEPS && w == 0) {
      tdm_load_tile(ldsA[cur ^ 1], Ab + (ks + 1) * 32, 32u, 128u, F_DIM, 3u, 3u);
      tdm_load_tile(ldsB[cur ^ 1], Bb + (ks + 1) * 32, 32u, 128u, F_DIM, 3u, 3u);
    }
    mma32x64(ldsA[cur], wm * 32, ldsB[cur], wn * 64, acc, lane);
  }
  int mg = blockIdx.x * 128 + wm * 32;
#pragma unroll
  for (int r = 0; r < 2; ++r) {
#pragma unroll
    for (int t = 0; t < 4; ++t) {
      int nbase = blockIdx.y * 128 + wn * 64 + t * 16 + (lane & 15);
      float bi = b2[nbase];
#pragma unroll
      for (int v = 0; v < 8; ++v) {
        int m = mg + r * 16 + v + ((lane & 16) ? 8 : 0);
        size_t idx = ((size_t)b * C_DIM + nbase) * N_POS + m;
        out[idx] = xres[idx] + acc[r][t][v] + bi;
      }
    }
  }
}

// =====================================================================
extern "C" void kernel_launch(void* const* d_in, const int* in_sizes, int n_in,
                              void* d_out, int out_size, void* d_ws, size_t ws_size,
                              hipStream_t stream) {
  (void)in_sizes; (void)n_in; (void)out_size; (void)ws_size;
  const float* src    = (const float*)d_in[0];
  const float* src1   = (const float*)d_in[1];
  const float* pos    = (const float*)d_in[2];
  const float* wq     = (const float*)d_in[3];
  const float* bq     = (const float*)d_in[4];
  const float* wk     = (const float*)d_in[5];
  const float* bk     = (const float*)d_in[6];
  const float* wv     = (const float*)d_in[7];
  const float* bv     = (const float*)d_in[8];
  const float* wo     = (const float*)d_in[9];
  const float* bo     = (const float*)d_in[10];
  const float* alpha1 = (const float*)d_in[11];
  const float* alpha2 = (const float*)d_in[12];
  const float* w1     = (const float*)d_in[13];
  const float* c1b    = (const float*)d_in[14];
  const float* w2     = (const float*)d_in[15];
  const float* c2b    = (const float*)d_in[16];
  float* out = (float*)d_out;

  size_t off = 0;
  char* ws = (char*)d_ws;
  auto alloc = [&](size_t bytes) -> char* {
    char* p = ws + off;
    off = (off + bytes + 255) & ~(size_t)255;
    return p;
  };
  const size_t BNC2 = (size_t)BATCH * N_POS * C_DIM * 2;   // bf16 [b][n][c]
  ushort_t* qin  = (ushort_t*)alloc(BNC2);
  ushort_t* kin  = (ushort_t*)alloc(BNC2);
  ushort_t* vin  = (ushort_t*)alloc(BNC2);
  ushort_t* wqb  = (ushort_t*)alloc((size_t)C_DIM * C_DIM * 2);
  ushort_t* wkb  = (ushort_t*)alloc((size_t)C_DIM * C_DIM * 2);
  ushort_t* wvb  = (ushort_t*)alloc((size_t)C_DIM * C_DIM * 2);
  ushort_t* wob  = (ushort_t*)alloc((size_t)C_DIM * C_DIM * 2);
  ushort_t* w1r  = (ushort_t*)alloc((size_t)9 * F_DIM * C_DIM * 2);
  ushort_t* w2b  = (ushort_t*)alloc((size_t)C_DIM * F_DIM * 2);
  ushort_t* Qb   = (ushort_t*)alloc(BNC2);
  ushort_t* Kb   = (ushort_t*)alloc(BNC2);
  ushort_t* Vtb  = (ushort_t*)alloc(BNC2);
  ushort_t* attnb= (ushort_t*)alloc(BNC2);
  float*    xres = (float*)   alloc((size_t)BATCH * C_DIM * N_POS * 4);
  ushort_t* xn2  = (ushort_t*)alloc(BNC2);
  ushort_t* yb   = (ushort_t*)alloc((size_t)BATCH * N_POS * F_DIM * 2);

  // weights -> bf16
  cvt_bf16_kernel<<<C_DIM * C_DIM / 256, 256, 0, stream>>>(wq, wqb, C_DIM * C_DIM);
  cvt_bf16_kernel<<<C_DIM * C_DIM / 256, 256, 0, stream>>>(wk, wkb, C_DIM * C_DIM);
  cvt_bf16_kernel<<<C_DIM * C_DIM / 256, 256, 0, stream>>>(wv, wvb, C_DIM * C_DIM);
  cvt_bf16_kernel<<<C_DIM * C_DIM / 256, 256, 0, stream>>>(wo, wob, C_DIM * C_DIM);
  cvt_bf16_kernel<<<C_DIM * F_DIM / 256, 256, 0, stream>>>(w2, w2b, C_DIM * F_DIM);
  repack_conv1_kernel<<<F_DIM * C_DIM / 256, 256, 0, stream>>>(w1, w1r);

  // norms + projections
  norm_qkv_kernel<<<BATCH * N_POS, 256, 0, stream>>>(src, src1, pos, alpha1, qin, kin, vin);
  dim3 gproj(N_POS / 128, C_DIM / 128, BATCH);
  proj_kernel<0><<<gproj, 256, 0, stream>>>(qin, wqb, bq, Qb, nullptr, nullptr);
  proj_kernel<1><<<gproj, 256, 0, stream>>>(kin, wkb, bk, Kb, nullptr, nullptr);
  proj_kernel<2><<<gproj, 256, 0, stream>>>(vin, wvb, bv, Vtb, nullptr, nullptr);

  // attention
  attn_kernel<<<dim3(N_POS / 64, BATCH * NHEAD), 256, 0, stream>>>(Qb, Kb, Vtb, attnb);

  // output projection + residual
  proj_kernel<3><<<gproj, 256, 0, stream>>>(attnb, wob, bo, nullptr, xres, src);

  // conv feed-forward
  norm2_kernel<<<BATCH * N_POS, 256, 0, stream>>>(xres, alpha2, xn2);
  conv1_kernel<<<dim3(N_POS / 128, F_DIM / 128, BATCH), 256, 0, stream>>>(xn2, w1r, c1b, yb);
  conv2_kernel<<<dim3(N_POS / 128, C_DIM / 128, BATCH), 256, 0, stream>>>(yb, w2b, c2b, xres, out);
}
